// DifuscoBackbone_1262720385541
// MI455X (gfx1250) — matrix operand order repaired and checked by reading, not compiled, not run on hardware
//
#include <hip/hip_runtime.h>
#include <hip/hip_bf16.h>

#define HIDDEN   128
#define LAYERS   4
#define N_NODES  10000
#define N_EDGES  160000
#define EPB      64    // edges per block (edge/head kernels)
#define NPB      80    // nodes per block (node gemm)

typedef __attribute__((ext_vector_type(16))) _Float16 v16h;
typedef __attribute__((ext_vector_type(8)))  _Float16 v8h;
typedef __attribute__((ext_vector_type(8)))  float    v8f;

// ---------------------------------------------------------------------------
// WMMA fragment helpers (V_WMMA_F32_16X16X32_F16, wave32)
//
// A (16x32 f16) per-lane layout (matched to B's k order):
//   lanes 0-15 : row M=lane,    K 0..15
//   lanes16-31 : row M=lane-16, K 16..31
// -> per lane: one contiguous 16-half (32B) run from a row-major f16 row.
//
// B (32x16 f16): lanes 0-15 col N=lane hold K 0..15 (2/VGPR), lanes 16-31
// hold K 16..31 -> per lane one contiguous 16-half (32B) run from a
// column-major (transposed, [n][k]) f16 weight.
//
// C/D (16x16 f32, 8 VGPRs): elem v -> row = v + 8*(lane>=16), col = lane&15.
// ---------------------------------------------------------------------------
__device__ __forceinline__ v16h load_a_frag(const _Float16* arow, int k0, int half) {
  v16h a = *(const v16h*)(arow + k0 + half * 16);
  return a;
}

__device__ __forceinline__ v8f wmma_acc(v16h a, v16h b, v8f c) {
  return __builtin_amdgcn_wmma_f32_16x16x32_f16(false, a, false, b, (short)0, c,
                                                false, false);
}

// Hoist the 4 B-fragments (full K=128) for one 16-wide column block.
__device__ __forceinline__ void load_b_frags(const _Float16* __restrict__ WT,
                                             int n0, int lr, int half, v16h b[4]) {
#pragma unroll
  for (int ks = 0; ks < 4; ++ks)
    b[ks] = *(const v16h*)(WT + (unsigned)((n0 + lr) * 128 + ks * 32 + half * 16));
}

// One 16x16 output tile: A row-tile (LDS f16, row-major) x hoisted B frags.
__device__ __forceinline__ v8f tile_gemm(const _Float16* arow, const v16h b[4],
                                         int half) {
  v8f acc = {};
#pragma unroll
  for (int ks = 0; ks < 4; ++ks)
    acc = wmma_acc(load_a_frag(arow, ks * 32, half), b[ks], acc);
  return acc;
}

// ---------------------------------------------------------------------------
// Weight conversion: f32 [m][k][n] -> f16 transposed [m][n][k]
// ---------------------------------------------------------------------------
__global__ void cvt_wT_kernel(const float* __restrict__ src,
                              _Float16* __restrict__ dst, int nmats) {
  int idx = blockIdx.x * 256 + threadIdx.x;
  if (idx >= nmats * 16384) return;
  int m = idx >> 14;
  int r = idx & 16383;
  int n = r >> 7, k = r & 127;
  dst[idx] = (_Float16)src[(m << 14) + k * 128 + n];
}

__global__ void zero_kernel(float* __restrict__ p, int n) {
  int i = blockIdx.x * 256 + threadIdx.x;
  if (i < n) p[i] = 0.f;
}

// ---------------------------------------------------------------------------
// Sine embeddings:  out[2k] = sin(x / 10000^(k/32)), out[2k+1] = cos(...)
// ---------------------------------------------------------------------------
__global__ void embed_nodes_kernel(const float* __restrict__ coords,
                                   float* __restrict__ h) {
  int idx = blockIdx.x * 256 + threadIdx.x;
  if (idx >= N_NODES * 128) return;
  int n = idx >> 7, c = idx & 127;
  float x = coords[n * 2 + (c >> 6)];
  int cc = c & 63, k = cc >> 1;
  float p = x * expf(-(float)k * (9.210340371976184f / 32.f));
  h[idx] = (cc & 1) ? cosf(p) : sinf(p);
}

__global__ void embed_edges_kernel(const float* __restrict__ dist,
                                   const float* __restrict__ x_t,
                                   float* __restrict__ e) {
  int idx = blockIdx.x * 256 + threadIdx.x;
  if (idx >= N_EDGES * 128) return;
  int i = idx >> 7, c = idx & 127;
  float x = (c < 64) ? dist[i] : x_t[i];
  int cc = c & 63, k = cc >> 1;
  float p = x * expf(-(float)k * (9.210340371976184f / 32.f));
  e[idx] = (cc & 1) ? cosf(p) : sinf(p);
}

// ---------------------------------------------------------------------------
// Time embedding + per-layer t_term (tiny: 1 block, 128 threads, VALU)
// ---------------------------------------------------------------------------
__global__ __launch_bounds__(128) void time_kernel(
    const float* __restrict__ t_in,
    const float* __restrict__ tw1, const float* __restrict__ tb1,
    const float* __restrict__ tw2, const float* __restrict__ tb2,
    const float* __restrict__ mw1, const float* __restrict__ mb1,
    const float* __restrict__ mw2, const float* __restrict__ mb2,
    float* __restrict__ tterm) {
  __shared__ float te[128], u[128], temb[128], r[128];
  const int j = threadIdx.x;
  const float tv = t_in[0];
  {
    int i = (j < 64) ? j : (j - 64);
    float a = tv * expf(-(float)i * (9.210340371976184f / 64.f));
    te[j] = (j < 64) ? cosf(a) : sinf(a);
  }
  __syncthreads();
  float a1 = tb1[j];
  for (int k = 0; k < 128; ++k) a1 += te[k] * tw1[k * 128 + j];
  u[j] = a1 / (1.f + expf(-a1));  // silu
  __syncthreads();
  float a2 = tb2[j];
  for (int k = 0; k < 128; ++k) a2 += u[k] * tw2[k * 128 + j];
  temb[j] = a2;
  __syncthreads();
  for (int l = 0; l < LAYERS; ++l) {
    float a3 = mb1[l * 128 + j];
    for (int k = 0; k < 128; ++k) a3 += temb[k] * mw1[l * 16384 + k * 128 + j];
    r[j] = a3 > 0.f ? a3 : 0.f;
    __syncthreads();
    float a4 = mb2[l * 128 + j];
    for (int k = 0; k < 128; ++k) a4 += r[k] * mw2[l * 16384 + k * 128 + j];
    tterm[l * 128 + j] = a4;
    __syncthreads();
  }
}

// ---------------------------------------------------------------------------
// Node GEMM: Qh/Rh/Vh/Uh = h @ {Q,R,V,U}.  NPB=80 rows/block, 5 row tiles.
// All 16 B-fragments (4 weights x K=128) hoisted; each A-fragment LDS load
// feeds 4 WMMAs (no duplicated ds traffic).
// ---------------------------------------------------------------------------
__global__ __launch_bounds__(256) void node_gemm_kernel(
    const float* __restrict__ h,
    const _Float16* __restrict__ Q16, const _Float16* __restrict__ R16,
    const _Float16* __restrict__ V16, const _Float16* __restrict__ U16,
    float* __restrict__ Qh, float* __restrict__ Rh,
    float* __restrict__ Vh, float* __restrict__ Uh) {
  __shared__ __align__(16) _Float16 sA[NPB][128];
  const int t = threadIdx.x;
  const int r0 = blockIdx.x * NPB;
  for (int idx = t; idx < NPB * 128; idx += 256) {
    int r = idx >> 7, c = idx & 127;
    sA[r][c] = (_Float16)h[(unsigned)((r0 + r) * 128 + c)];
  }
  __syncthreads();
  const int lane = t & 31, half = lane >> 4, lr = lane & 15;
  const int n0 = (t >> 5) * 16;

  v16h bq[4], br[4], bv[4], bu[4];
  load_b_frags(Q16, n0, lr, half, bq);
  load_b_frags(R16, n0, lr, half, br);
  load_b_frags(V16, n0, lr, half, bv);
  load_b_frags(U16, n0, lr, half, bu);

#pragma unroll
  for (int rt = 0; rt < NPB / 16; ++rt) {
    v8f aq = {}, ar = {}, av = {}, au = {};
    const _Float16* arow = &sA[rt * 16 + lr][0];
#pragma unroll
    for (int ks = 0; ks < 4; ++ks) {
      v16h a = load_a_frag(arow, ks * 32, half);
      aq = wmma_acc(a, bq[ks], aq);
      ar = wmma_acc(a, br[ks], ar);
      av = wmma_acc(a, bv[ks], av);
      au = wmma_acc(a, bu[ks], au);
    }
#pragma unroll
    for (int v = 0; v < 8; ++v) {
      const int row = rt * 16 + v + half * 8, col = n0 + lr;
      const unsigned o = (unsigned)((r0 + row) * 128 + col);
      Qh[o] = aq[v]; Rh[o] = ar[v]; Vh[o] = av[v]; Uh[o] = au[v];
    }
  }
}

// ---------------------------------------------------------------------------
// Fused edge layer (EPB=64 edges/block, 8 waves; wave = one 16-col block,
// 4 row tiles; B-fragments hoisted per stage and reused 4x):
//   e_hat = e@P + Qh[src] + Rh[dst];  en = LN(e_hat)
//   e    += relu(en@W1 + b1)@W2 + b2 + t_term          (in-place residual)
//   agg[src] += sigmoid(e_hat) * Vh[dst]               (f32 global atomics)
// ---------------------------------------------------------------------------
__global__ __launch_bounds__(256) void edge_layer_kernel(
    float* __restrict__ e, const int* __restrict__ srcIdx,
    const int* __restrict__ dstIdx, const float* __restrict__ Qh,
    const float* __restrict__ Rh, const float* __restrict__ Vh,
    const _Float16* __restrict__ P16, const _Float16* __restrict__ W116,
    const _Float16* __restrict__ W216, const float* __restrict__ mb1,
    const float* __restrict__ mb2, const float* __restrict__ lng,
    const float* __restrict__ lnb, const float* __restrict__ tterm,
    float* __restrict__ agg) {
  __shared__ float sH[EPB][128];                    // e_hat (32KB)
  __shared__ __align__(16) _Float16 sA[EPB][128];   // GEMM A input (16KB)
  __shared__ __align__(16) _Float16 sB[EPB][128];   // MLP hidden (16KB)
  __shared__ float rs[EPB][4], rq[EPB][4];
  __shared__ int sSrc[EPB], sDst[EPB];

  const int t = threadIdx.x;
  const int e0 = blockIdx.x * EPB;

  for (int idx = t; idx < EPB * 128; idx += 256) {
    int r = idx >> 7, c = idx & 127;
    sA[r][c] = (_Float16)e[(unsigned)((e0 + r) * 128 + c)];
  }
  if (t < EPB) { sSrc[t] = srcIdx[e0 + t] * 128; sDst[t] = dstIdx[e0 + t] * 128; }
  __syncthreads();

  const int lane = t & 31, half = lane >> 4, lr = lane & 15;
  const int n0 = (t >> 5) * 16;
  const int col = n0 + lr;

  // ---- stage 1: e_hat = e@P + gather(Qh,src) + gather(Rh,dst) -> sH ----
  {
    v16h b[4];
    load_b_frags(P16, n0, lr, half, b);
#pragma unroll
    for (int rt = 0; rt < EPB / 16; ++rt) {
      v8f acc = tile_gemm(&sA[rt * 16 + lr][0], b, half);
#pragma unroll
      for (int v = 0; v < 8; ++v) {
        const int row = rt * 16 + v + half * 8;
        sH[row][col] = acc[v] + Qh[(unsigned)(sSrc[row] + col)] +
                       Rh[(unsigned)(sDst[row] + col)];
      }
    }
  }
  __syncthreads();

  // ---- layer_norm(e_hat) -> sA (f16), 4 threads per row ----
  {
    const int row = t >> 2, j = t & 3;
    float s = 0.f, q = 0.f;
    for (int c = j; c < 128; c += 4) { float x = sH[row][c]; s += x; q += x * x; }
    rs[row][j] = s; rq[row][j] = q;
    __syncthreads();
    float mu = 0.f, m2 = 0.f;
#pragma unroll
    for (int jj = 0; jj < 4; ++jj) { mu += rs[row][jj]; m2 += rq[row][jj]; }
    mu *= (1.f / 128.f); m2 *= (1.f / 128.f);
    const float rstd = rsqrtf(m2 - mu * mu + 1e-5f);
    for (int c = j; c < 128; c += 4)
      sA[row][c] = (_Float16)((sH[row][c] - mu) * rstd * lng[c] + lnb[c]);
  }
  __syncthreads();

  // ---- stage 2: m1 = relu(en@W1 + b1) -> sB (f16) ----
  {
    v16h b[4];
    load_b_frags(W116, n0, lr, half, b);
    const float bias1 = mb1[col];
#pragma unroll
    for (int rt = 0; rt < EPB / 16; ++rt) {
      v8f acc = tile_gemm(&sA[rt * 16 + lr][0], b, half);
#pragma unroll
      for (int v = 0; v < 8; ++v) {
        const int row = rt * 16 + v + half * 8;
        const float x = acc[v] + bias1;
        sB[row][col] = (_Float16)(x > 0.f ? x : 0.f);
      }
    }
  }
  __syncthreads();

  // ---- stage 3: e += m1@W2 + b2 + t_term ; scatter gated messages ----
  {
    v16h b[4];
    load_b_frags(W216, n0, lr, half, b);
    const float bias2 = mb2[col] + tterm[col];
#pragma unroll
    for (int rt = 0; rt < EPB / 16; ++rt) {
      v8f acc = tile_gemm(&sB[rt * 16 + lr][0], b, half);
#pragma unroll
      for (int v = 0; v < 8; ++v) {
        const int row = rt * 16 + v + half * 8;
        const unsigned o = (unsigned)((e0 + row) * 128 + col);
        e[o] = e[o] + acc[v] + bias2;  // residual: re-read original e (L2 hit)
        const float g = 1.f / (1.f + expf(-sH[row][col]));
        const float m = g * Vh[(unsigned)(sDst[row] + col)];
        atomicAdd(&agg[(unsigned)(sSrc[row] + col)], m);
      }
    }
  }
}

// ---------------------------------------------------------------------------
// h += relu(layer_norm(Uh + agg));  one node per 128-thread block
// ---------------------------------------------------------------------------
__global__ __launch_bounds__(128) void node_update_kernel(
    float* __restrict__ h, const float* __restrict__ Uh,
    const float* __restrict__ agg, const float* __restrict__ g,
    const float* __restrict__ b) {
  __shared__ float rs[128], rq[128];
  const int n = blockIdx.x, c = threadIdx.x;
  const unsigned o = (unsigned)(n * 128 + c);
  const float x = Uh[o] + agg[o];
  rs[c] = x; rq[c] = x * x;
  __syncthreads();
  for (int s = 64; s > 0; s >>= 1) {
    if (c < s) { rs[c] += rs[c + s]; rq[c] += rq[c + s]; }
    __syncthreads();
  }
  const float mu = rs[0] * (1.f / 128.f);
  const float rstd = rsqrtf(rq[0] * (1.f / 128.f) - mu * mu + 1e-5f);
  const float y = (x - mu) * rstd * g[c] + b[c];
  h[o] += (y > 0.f ? y : 0.f);
}

// ---------------------------------------------------------------------------
// Head: out = relu(e@W1 + b1) @ W2 + b2     (W2 is 128x2; scalar tail)
// ---------------------------------------------------------------------------
__global__ __launch_bounds__(256) void head_kernel(
    const float* __restrict__ e, const _Float16* __restrict__ W116,
    const float* __restrict__ b1, const float* __restrict__ w2,
    const float* __restrict__ b2, float* __restrict__ out) {
  __shared__ __align__(16) _Float16 sA[EPB][128];
  __shared__ float sM[EPB][128];
  const int t = threadIdx.x;
  const int e0 = blockIdx.x * EPB;
  for (int idx = t; idx < EPB * 128; idx += 256) {
    int r = idx >> 7, c = idx & 127;
    sA[r][c] = (_Float16)e[(unsigned)((e0 + r) * 128 + c)];
  }
  __syncthreads();
  const int lane = t & 31, half = lane >> 4, lr = lane & 15;
  const int n0 = (t >> 5) * 16;
  const int col = n0 + lr;
  {
    v16h b[4];
    load_b_frags(W116, n0, lr, half, b);
    const float bias1 = b1[col];
#pragma unroll
    for (int rt = 0; rt < EPB / 16; ++rt) {
      v8f acc = tile_gemm(&sA[rt * 16 + lr][0], b, half);
#pragma unroll
      for (int v = 0; v < 8; ++v) {
        const int row = rt * 16 + v + half * 8;
        const float x = acc[v] + bias1;
        sM[row][col] = x > 0.f ? x : 0.f;
      }
    }
  }
  __syncthreads();
  if (t < 2 * EPB) {
    const int row = t >> 1, j = t & 1;
    float s = b2[j];
    for (int k = 0; k < 128; ++k) s += sM[row][k] * w2[k * 2 + j];
    out[(unsigned)((e0 + row) * 2 + j)] = s;
  }
}

// ---------------------------------------------------------------------------
extern "C" void kernel_launch(void* const* d_in, const int* in_sizes, int n_in,
                              void* d_out, int out_size, void* d_ws,
                              size_t ws_size, hipStream_t stream) {
  (void)n_in; (void)out_size; (void)ws_size;

  // Input flattening: support both insertion-order and JAX sorted-key order.
  const float *node_coords, *edge_dist, *x_t, *t_in;
  const int* edge_index;
  const float *P, *Q, *R, *U, *V, *ln_e_g, *ln_e_b, *ln_h_g, *ln_h_b;
  const float *emlp_w1, *emlp_b1, *emlp_w2, *emlp_b2;
  const float *tmlp_w1, *tmlp_b1, *tmlp_w2, *tmlp_b2;
  const float *tp_w1, *tp_b1, *tp_w2, *tp_b2;
  const float *eh_w1, *eh_b1, *eh_w2, *eh_b2;

  if (in_sizes[0] == N_NODES * 2) {
    // insertion order: node_coords, edge_index, edge_distances, x_t, t, params{...}
    node_coords = (const float*)d_in[0];
    edge_index  = (const int*)d_in[1];
    edge_dist   = (const float*)d_in[2];
    x_t         = (const float*)d_in[3];
    t_in        = (const float*)d_in[4];
    P = (const float*)d_in[5];  Q = (const float*)d_in[6];
    R = (const float*)d_in[7];  U = (const float*)d_in[8];
    V = (const float*)d_in[9];
    ln_e_g = (const float*)d_in[10]; ln_e_b = (const float*)d_in[11];
    ln_h_g = (const float*)d_in[12]; ln_h_b = (const float*)d_in[13];
    emlp_w1 = (const float*)d_in[14]; emlp_b1 = (const float*)d_in[15];
    emlp_w2 = (const float*)d_in[16]; emlp_b2 = (const float*)d_in[17];
    tmlp_w1 = (const float*)d_in[18]; tmlp_b1 = (const float*)d_in[19];
    tmlp_w2 = (const float*)d_in[20]; tmlp_b2 = (const float*)d_in[21];
    tp_w1 = (const float*)d_in[22]; tp_b1 = (const float*)d_in[23];
    tp_w2 = (const float*)d_in[24]; tp_b2 = (const float*)d_in[25];
    eh_w1 = (const float*)d_in[26]; eh_b1 = (const float*)d_in[27];
    eh_w2 = (const float*)d_in[28]; eh_b2 = (const float*)d_in[29];
  } else {
    // jax.tree_flatten sorted-key order
    edge_dist   = (const float*)d_in[0];
    edge_index  = (const int*)d_in[1];
    node_coords = (const float*)d_in[2];
    eh_b1 = (const float*)d_in[3];  eh_b2 = (const float*)d_in[4];
    eh_w1 = (const float*)d_in[5];  eh_w2 = (const float*)d_in[6];
    P = (const float*)d_in[7];  Q = (const float*)d_in[8];
    R = (const float*)d_in[9];  U = (const float*)d_in[10];
    V = (const float*)d_in[11];
    emlp_b1 = (const float*)d_in[12]; emlp_b2 = (const float*)d_in[13];
    emlp_w1 = (const float*)d_in[14]; emlp_w2 = (const float*)d_in[15];
    ln_e_b = (const float*)d_in[16]; ln_e_g = (const float*)d_in[17];
    ln_h_b = (const float*)d_in[18]; ln_h_g = (const float*)d_in[19];
    tmlp_b1 = (const float*)d_in[20]; tmlp_b2 = (const float*)d_in[21];
    tmlp_w1 = (const float*)d_in[22]; tmlp_w2 = (const float*)d_in[23];
    tp_b1 = (const float*)d_in[24]; tp_b2 = (const float*)d_in[25];
    tp_w1 = (const float*)d_in[26]; tp_w2 = (const float*)d_in[27];
    t_in = (const float*)d_in[28];
    x_t  = (const float*)d_in[29];
  }

  const int* srcIdx = edge_index;
  const int* dstIdx = edge_index + N_EDGES;

  // Workspace carve (256B-aligned). Total ~114 MB.
  char* wp = (char*)d_ws;
  auto carve = [&](size_t bytes) -> void* {
    void* r = (void*)wp;
    wp += (bytes + 255) & ~(size_t)255;
    return r;
  };
  float* e   = (float*)carve((size_t)N_EDGES * 128 * 4);
  float* h   = (float*)carve((size_t)N_NODES * 128 * 4);
  float* Qh  = (float*)carve((size_t)N_NODES * 128 * 4);
  float* Rh  = (float*)carve((size_t)N_NODES * 128 * 4);
  float* Vh  = (float*)carve((size_t)N_NODES * 128 * 4);
  float* Uh  = (float*)carve((size_t)N_NODES * 128 * 4);
  float* agg = (float*)carve((size_t)N_NODES * 128 * 4);
  float* tterm = (float*)carve((size_t)LAYERS * 128 * 4);
  _Float16* P16  = (_Float16*)carve((size_t)LAYERS * 16384 * 2);
  _Float16* Q16  = (_Float16*)carve((size_t)LAYERS * 16384 * 2);
  _Float16* R16  = (_Float16*)carve((size_t)LAYERS * 16384 * 2);
  _Float16* U16  = (_Float16*)carve((size_t)LAYERS * 16384 * 2);
  _Float16* V16  = (_Float16*)carve((size_t)LAYERS * 16384 * 2);
  _Float16* EW1  = (_Float16*)carve((size_t)LAYERS * 16384 * 2);
  _Float16* EW2  = (_Float16*)carve((size_t)LAYERS * 16384 * 2);
  _Float16* HW1  = (_Float16*)carve((size_t)16384 * 2);

  // Per-launch deterministic weight convert+transpose to f16.
  const int cvB = (LAYERS * 16384 + 255) / 256;
  cvt_wT_kernel<<<cvB, 256, 0, stream>>>(P, P16, LAYERS);
  cvt_wT_kernel<<<cvB, 256, 0, stream>>>(Q, Q16, LAYERS);
  cvt_wT_kernel<<<cvB, 256, 0, stream>>>(R, R16, LAYERS);
  cvt_wT_kernel<<<cvB, 256, 0, stream>>>(U, U16, LAYERS);
  cvt_wT_kernel<<<cvB, 256, 0, stream>>>(V, V16, LAYERS);
  cvt_wT_kernel<<<cvB, 256, 0, stream>>>(emlp_w1, EW1, LAYERS);
  cvt_wT_kernel<<<cvB, 256, 0, stream>>>(emlp_w2, EW2, LAYERS);
  cvt_wT_kernel<<<64, 256, 0, stream>>>(eh_w1, HW1, 1);

  embed_nodes_kernel<<<(N_NODES * 128) / 256, 256, 0, stream>>>(node_coords, h);
  embed_edges_kernel<<<(N_EDGES * 128) / 256, 256, 0, stream>>>(edge_dist, x_t, e);
  time_kernel<<<1, 128, 0, stream>>>(t_in, tp_w1, tp_b1, tp_w2, tp_b2,
                                     tmlp_w1, tmlp_b1, tmlp_w2, tmlp_b2, tterm);

  for (int l = 0; l < LAYERS; ++l) {
    zero_kernel<<<(N_NODES * 128) / 256, 256, 0, stream>>>(agg, N_NODES * 128);
    node_gemm_kernel<<<N_NODES / NPB, 256, 0, stream>>>(
        h, Q16 + l * 16384, R16 + l * 16384, V16 + l * 16384, U16 + l * 16384,
        Qh, Rh, Vh, Uh);
    edge_layer_kernel<<<N_EDGES / EPB, 256, 0, stream>>>(
        e, srcIdx, dstIdx, Qh, Rh, Vh, P16 + l * 16384, EW1 + l * 16384,
        EW2 + l * 16384, emlp_b1 + l * 128, emlp_b2 + l * 128,
        ln_e_g + l * 128, ln_e_b + l * 128, tterm + l * 128, agg);
    node_update_kernel<<<N_NODES, 128, 0, stream>>>(
        h, Uh, agg, ln_h_g + l * 128, ln_h_b + l * 128);
  }
  head_kernel<<<N_EDGES / EPB, 256, 0, stream>>>(e, HW1, eh_b1, eh_w2, eh_b2,
                                                 (float*)d_out);
}
